// TokenImportancePredictorAttentive_20469814133365
// MI455X (gfx1250) — matrix operand order repaired and checked by reading, not compiled, not run on hardware
//
#include <hip/hip_runtime.h>

typedef unsigned short u16;
typedef unsigned int u32;
typedef __attribute__((ext_vector_type(16))) __bf16 v16bf;
typedef __attribute__((ext_vector_type(8)))  float  v8f;

union Frag { v16bf v; uint4 q[2]; };

static constexpr int L_SEQ  = 2048;
static constexpr int E_DIM  = 4096;
static constexpr int HR_DIM = 1024;

#if defined(__has_builtin)
#if __has_builtin(__builtin_amdgcn_global_load_async_to_lds_b128) && \
    __has_builtin(__builtin_amdgcn_s_wait_asynccnt)
#define HAS_ASYNC_LDS 1
#endif
#endif

#ifdef HAS_ASYNC_LDS
typedef int v4i __attribute__((ext_vector_type(4)));
typedef __attribute__((address_space(1))) v4i gv4i;   // global AS pointer elem
typedef __attribute__((address_space(3))) v4i lv4i;   // LDS AS pointer elem
__device__ __forceinline__ void async_cp16(const u16* g, u16* l) {
  // GLOBAL_LOAD_ASYNC_TO_LDS_B128: 16 bytes per lane, tracked by ASYNCcnt
  __builtin_amdgcn_global_load_async_to_lds_b128((gv4i*)g, (lv4i*)l, 0, 0);
}
#endif

__device__ __forceinline__ u16 f2bf(float f) {
  u32 u = __float_as_uint(f);
  u += 0x7FFFu + ((u >> 16) & 1u);        // round to nearest even
  return (u16)(u >> 16);
}
__device__ __forceinline__ float bf2f(u16 h) {
  return __uint_as_float(((u32)h) << 16);
}

__device__ __forceinline__ v8f wmma_bf16(Frag a, Frag b, v8f c) {
  return __builtin_amdgcn_wmma_f32_16x16x32_bf16(false, a.v, false, b.v,
                                                 (short)0, c, false, false);
}

// ---------------------------------------------------------------- conversion
__global__ void __launch_bounds__(256) cvt_f32_bf16(const float* __restrict__ s,
                                                    u16* __restrict__ d, int n4) {
  int i = blockIdx.x * 256 + threadIdx.x;
  if (i < n4) {
    float4 f = ((const float4*)s)[i];
    ushort4 o;
    o.x = f2bf(f.x); o.y = f2bf(f.y); o.z = f2bf(f.z); o.w = f2bf(f.w);
    ((ushort4*)d)[i] = o;
  }
}

// ---------------------------------------------------------------- GEMM
// C[M,N] = A[M,K] (bf16, row-major) x B[N,K]^T (bf16, row-major weights)
enum { EPI_BF16 = 0, EPI_BIAS_GELU = 1, EPI_SILU = 2, EPI_BIAS_RESID = 3,
       EPI_T = 4, EPI_ROPE = 5 };

template <int EPI>
__global__ void __launch_bounds__(256) gemm_bf16k(
    const u16* __restrict__ A, const u16* __restrict__ B,
    const float* __restrict__ bias, const float* __restrict__ resid,
    void* __restrict__ Out, int M, int N, int K)
{
  constexpr int BM = 128, BN = 128, BK = 32, LDT = 40;   // padded LDS stride
#ifdef HAS_ASYNC_LDS
  __shared__ __align__(16) u16 sA[2][BM * LDT];          // double-buffered
  __shared__ __align__(16) u16 sB[2][BN * LDT];
#else
  __shared__ __align__(16) u16 sA[1][BM * LDT];
  __shared__ __align__(16) u16 sB[1][BN * LDT];
#endif
  const int tid  = threadIdx.x;
  const int lane = tid & 31, wid = tid >> 5;
  const int wm = wid >> 2, wn = wid & 3;                 // 2x4 wave grid
  const int lrow = lane & 15, kh = lane >> 4;
  const int bm0 = blockIdx.y * BM, bn0 = blockIdx.x * BN;

  v8f zero = {};
  v8f acc[4][2];
#pragma unroll
  for (int m = 0; m < 4; ++m) { acc[m][0] = zero; acc[m][1] = zero; }

  const int ldRow = tid >> 1;
  const int ldCol = (tid & 1) * 16;
  const u16* Ag = A + (size_t)(bm0 + ldRow) * K + ldCol;
  const u16* Bg = B + (size_t)(bn0 + ldRow) * K + ldCol;
  const int woff = ldRow * LDT + ldCol;

  auto compute_tile = [&](const u16* cA, const u16* cB) {
    Frag fa[4], fb[2];
#pragma unroll
    for (int mt = 0; mt < 4; ++mt) {   // A frag: K = {8*kh..}, {16+8*kh..}
      const u16* p = cA + (wm * 64 + mt * 16 + lrow) * LDT + kh * 8;
      fa[mt].q[0] = *(const uint4*)(p);
      fa[mt].q[1] = *(const uint4*)(p + 16);
    }
#pragma unroll
    for (int nt = 0; nt < 2; ++nt) {   // B frag: K = {8*kh..}, {16+8*kh..}
      const u16* p = cB + (wn * 32 + nt * 16 + lrow) * LDT + kh * 8;
      fb[nt].q[0] = *(const uint4*)(p);
      fb[nt].q[1] = *(const uint4*)(p + 16);
    }
#pragma unroll
    for (int mt = 0; mt < 4; ++mt)
#pragma unroll
      for (int nt = 0; nt < 2; ++nt)
        acc[mt][nt] = wmma_bf16(fa[mt], fb[nt], acc[mt][nt]);
  };

#ifdef HAS_ASYNC_LDS
  const int nk = K / BK;
  // prologue: tile 0 in flight
  async_cp16(Ag,     sA[0] + woff);
  async_cp16(Ag + 8, sA[0] + woff + 8);
  async_cp16(Bg,     sB[0] + woff);
  async_cp16(Bg + 8, sB[0] + woff + 8);
  Ag += BK; Bg += BK;
  for (int kb = 0; kb < nk; ++kb) {
    const int cur = kb & 1;
    if (kb + 1 < nk) {
      const int nxt = cur ^ 1;       // prefetch tile kb+1 while kb computes
      async_cp16(Ag,     sA[nxt] + woff);
      async_cp16(Ag + 8, sA[nxt] + woff + 8);
      async_cp16(Bg,     sB[nxt] + woff);
      async_cp16(Bg + 8, sB[nxt] + woff + 8);
      Ag += BK; Bg += BK;
      __builtin_amdgcn_s_wait_asynccnt(4);  // oldest 4 (tile kb) landed
    } else {
      __builtin_amdgcn_s_wait_asynccnt(0);
    }
    __syncthreads();
    compute_tile(sA[cur], sB[cur]);
    __syncthreads();
  }
#else
  for (int k0 = 0; k0 < K; k0 += BK) {
    uint4 a0 = *(const uint4*)(Ag);
    uint4 a1 = *(const uint4*)(Ag + 8);
    uint4 b0 = *(const uint4*)(Bg);
    uint4 b1 = *(const uint4*)(Bg + 8);
    Ag += BK; Bg += BK;
    __syncthreads();
    *(uint4*)(sA[0] + woff)     = a0; *(uint4*)(sA[0] + woff + 8) = a1;
    *(uint4*)(sB[0] + woff)     = b0; *(uint4*)(sB[0] + woff + 8) = b1;
    __syncthreads();
    compute_tile(sA[0], sB[0]);
  }
#endif

#pragma unroll
  for (int mt = 0; mt < 4; ++mt) {
#pragma unroll
    for (int nt = 0; nt < 2; ++nt) {
      const int rb   = bm0 + wm * 64 + mt * 16 + kh * 8;
      const int gcol = bn0 + wn * 32 + nt * 16 + lrow;
#pragma unroll
      for (int i = 0; i < 8; ++i) {
        float v = acc[mt][nt][i];
        const int grow = rb + i;
        if (EPI == EPI_BF16) {
          ((u16*)Out)[(size_t)grow * N + gcol] = f2bf(v);
        } else if (EPI == EPI_BIAS_GELU) {
          v += bias[gcol];
          v = 0.5f * v * (1.0f + erff(v * 0.70710678118654752f));
          ((u16*)Out)[(size_t)grow * N + gcol] = f2bf(v);
        } else if (EPI == EPI_SILU) {
          v = v / (1.0f + __expf(-v));
          ((u16*)Out)[(size_t)grow * N + gcol] = f2bf(v);
        } else if (EPI == EPI_BIAS_RESID) {
          v += bias[gcol] + resid[(size_t)grow * N + gcol];
          ((u16*)Out)[(size_t)grow * N + gcol] = f2bf(v);
        } else if (EPI == EPI_T) {
          ((u16*)Out)[(size_t)gcol * M + grow] = f2bf(v);
        } else {  // EPI_ROPE: fused rope(d=16) + (B,N,H,L,D) permuted fp32 store
          float pv = __shfl_xor(v, 8, 32);       // partner column dd^8
          const int dd = gcol & 15;
          const int fi = dd & 7;
          const float inv = expf(-(float)fi * 0.125f * 9.210340371976184f);
          const float ang = (float)grow * inv;
          const float c = cosf(ang), s = sinf(ang);
          const float val = (dd < 8) ? (v * c - pv * s) : (v * c + pv * s);
          const int nn = gcol >> 9;
          const int hh = (gcol >> 4) & 31;
          const size_t oidx =
              (((size_t)(nn * 32 + hh)) * (size_t)L_SEQ + (size_t)grow) * 16 + dd;
          ((float*)Out)[oidx] = val;
        }
      }
    }
  }
}

// ---------------------------------------------------------------- RoPE d=32 on Q/K
__global__ void __launch_bounds__(256) rope32_kernel(u16* __restrict__ buf) {
  const int t = blockIdx.x * 256 + threadIdx.x;     // L * H * 16 threads
  const int l = t >> 9;
  const int r = t & 511;
  const int h = r >> 4, f = r & 15;
  const size_t base = (size_t)l * HR_DIM + h * 32 + f;
  const float inv = expf(-(float)f * (1.0f / 16.0f) * 9.210340371976184f);
  const float ang = (float)l * inv;
  const float c = cosf(ang), s = sinf(ang);
  const float x0 = bf2f(buf[base]), x1 = bf2f(buf[base + 16]);
  buf[base]      = f2bf(x0 * c - x1 * s);
  buf[base + 16] = f2bf(x1 * c + x0 * s);
}

// ---------------------------------------------------------------- flash attention
// grid (16 qblocks, 32 heads), 256 threads; wave w owns 16 query rows.
__global__ void __launch_bounds__(256) attn_kernel(
    const u16* __restrict__ Q, const u16* __restrict__ Kb,
    const u16* __restrict__ Vt, u16* __restrict__ O)
{
  constexpr int LDT = 40;
  __shared__ __align__(16) u16 sK[32 * LDT];
  __shared__ __align__(16) u16 sV[32 * LDT];
  __shared__ __align__(16) u16 sP[8 * 16 * LDT];
  const int h = blockIdx.y, qb = blockIdx.x;
  const int tid = threadIdx.x, lane = tid & 31, w = tid >> 5;
  const int lrow = lane & 15, kh = lane >> 4;
  const int qr0 = qb * 128 + w * 16;

  Frag fq;
  {
    const u16* p = Q + (size_t)(qr0 + lrow) * HR_DIM + h * 32 + kh * 16;
    fq.q[0] = *(const uint4*)(p);
    fq.q[1] = *(const uint4*)(p + 8);
  }
  v8f zero = {};
  v8f o0 = zero, o1 = zero;
  float mrow[8], lsum[8];
#pragma unroll
  for (int i = 0; i < 8; ++i) { mrow[i] = -1e30f; lsum[i] = 0.0f; }

  const float scale = 0.17677669529663687f;   // 1/sqrt(32)
  const int nkb = qb * 4 + 4;
  const int lr = tid >> 3;
  const int lc = (tid & 7) * 4;

  for (int kb = 0; kb < nkb; ++kb) {
    const int k0 = kb * 32;
    uint2 kv = *(const uint2*)(Kb + (size_t)(k0 + lr) * HR_DIM + h * 32 + lc);
    uint2 vv = *(const uint2*)(Vt + (size_t)(h * 32 + lr) * L_SEQ + k0 + lc);
    __syncthreads();
    *(uint2*)(sK + lr * LDT + lc) = kv;
    *(uint2*)(sV + lr * LDT + lc) = vv;
    __syncthreads();

    if (k0 <= qr0 + 15) {                       // wave-uniform causal skip
      Frag fk0, fk1, fv0, fv1;
      {
        const u16* p0 = sK + lrow * LDT + kh * 16;
        fk0.q[0] = *(const uint4*)(p0); fk0.q[1] = *(const uint4*)(p0 + 8);
        const u16* p1 = sK + (16 + lrow) * LDT + kh * 16;
        fk1.q[0] = *(const uint4*)(p1); fk1.q[1] = *(const uint4*)(p1 + 8);
        const u16* p2 = sV + lrow * LDT + kh * 16;
        fv0.q[0] = *(const uint4*)(p2); fv0.q[1] = *(const uint4*)(p2 + 8);
        const u16* p3 = sV + (16 + lrow) * LDT + kh * 16;
        fv1.q[0] = *(const uint4*)(p3); fv1.q[1] = *(const uint4*)(p3 + 8);
      }
      v8f s0 = wmma_bf16(fq, fk0, zero);
      v8f s1 = wmma_bf16(fq, fk1, zero);
      u16* sp = sP + w * 16 * LDT;
      float alpha[8];
#pragma unroll
      for (int i = 0; i < 8; ++i) {
        const int row  = qr0 + i + 8 * kh;
        const int key0 = k0 + lrow;
        float a = s0[i] * scale; if (key0 > row)      a = -1e30f;
        float b = s1[i] * scale; if (key0 + 16 > row) b = -1e30f;
        float mx = fmaxf(a, b);
#pragma unroll
        for (int off = 1; off < 16; off <<= 1) mx = fmaxf(mx, __shfl_xor(mx, off, 32));
        const float mnew = fmaxf(mrow[i], mx);
        const float al = __expf(mrow[i] - mnew);
        const float e0 = __expf(a - mnew);
        const float e1 = __expf(b - mnew);
        float rs = e0 + e1;
#pragma unroll
        for (int off = 1; off < 16; off <<= 1) rs += __shfl_xor(rs, off, 32);
        lsum[i] = lsum[i] * al + rs;
        mrow[i] = mnew;
        alpha[i] = al;
        sp[(i + 8 * kh) * LDT + lrow]      = f2bf(e0);   // C-layout -> LDS
        sp[(i + 8 * kh) * LDT + 16 + lrow] = f2bf(e1);
      }
#pragma unroll
      for (int i = 0; i < 8; ++i) { o0[i] *= alpha[i]; o1[i] *= alpha[i]; }
      Frag fp;                                           // reload in A-layout
      const u16* pp = sp + lrow * LDT + kh * 16;
      fp.q[0] = *(const uint4*)(pp);
      fp.q[1] = *(const uint4*)(pp + 8);
      o0 = wmma_bf16(fp, fv0, o0);
      o1 = wmma_bf16(fp, fv1, o1);
    }
  }
#pragma unroll
  for (int i = 0; i < 8; ++i) {
    const float inv = 1.0f / lsum[i];
    const int row = qr0 + i + 8 * kh;
    O[(size_t)row * HR_DIM + h * 32 + lrow]      = f2bf(o0[i] * inv);
    O[(size_t)row * HR_DIM + h * 32 + 16 + lrow] = f2bf(o1[i] * inv);
  }
}

// ---------------------------------------------------------------- LayerNorm
__global__ void __launch_bounds__(256) ln_kernel(
    const u16* __restrict__ X, const float* __restrict__ wgt,
    const float* __restrict__ bia, u16* __restrict__ Y)
{
  __shared__ float ssum[8], ssq[8];
  const int row = blockIdx.x, tid = threadIdx.x;
  float v[4], sum = 0.f, sq = 0.f;
#pragma unroll
  for (int t = 0; t < 4; ++t) {
    float x = bf2f(X[(size_t)row * HR_DIM + tid + t * 256]);
    v[t] = x; sum += x; sq += x * x;
  }
#pragma unroll
  for (int off = 16; off; off >>= 1) {
    sum += __shfl_xor(sum, off, 32);
    sq  += __shfl_xor(sq,  off, 32);
  }
  if ((tid & 31) == 0) { ssum[tid >> 5] = sum; ssq[tid >> 5] = sq; }
  __syncthreads();
  float ts = 0.f, tq = 0.f;
#pragma unroll
  for (int i = 0; i < 8; ++i) { ts += ssum[i]; tq += ssq[i]; }
  const float mean = ts * (1.0f / 1024.0f);
  const float var  = tq * (1.0f / 1024.0f) - mean * mean;
  const float rstd = rsqrtf(var + 1e-5f);
#pragma unroll
  for (int t = 0; t < 4; ++t) {
    int j = tid + t * 256;
    Y[(size_t)row * HR_DIM + j] = f2bf((v[t] - mean) * rstd * wgt[j] + bia[j]);
  }
}

// ---------------------------------------------------------------- host
extern "C" void kernel_launch(void* const* d_in, const int* in_sizes, int n_in,
                              void* d_out, int out_size, void* d_ws, size_t ws_size,
                              hipStream_t stream) {
  (void)in_sizes; (void)n_in; (void)out_size; (void)ws_size;
  const float* x    = (const float*)d_in[0];
  const float* Wi   = (const float*)d_in[1];
  const float* Wq   = (const float*)d_in[2];
  const float* Wk   = (const float*)d_in[3];
  const float* Wv   = (const float*)d_in[4];
  const float* ln1w = (const float*)d_in[5];
  const float* ln1b = (const float*)d_in[6];
  const float* Wf1  = (const float*)d_in[7];
  const float* bf1  = (const float*)d_in[8];
  const float* Wf2  = (const float*)d_in[9];
  const float* bf2v = (const float*)d_in[10];
  const float* Wqi1 = (const float*)d_in[11];
  const float* Wqi2 = (const float*)d_in[12];
  const float* Wki1 = (const float*)d_in[13];
  const float* Wki2 = (const float*)d_in[14];
  float* out = (float*)d_out;

  char* wsp = (char*)d_ws;
  auto alloc = [&](size_t bytes) -> void* {
    void* p = wsp; wsp += (bytes + 255) & ~(size_t)255; return p;
  };
  u16* xb    = (u16*)alloc((size_t)L_SEQ * E_DIM * 2);
  u16* wib   = (u16*)alloc((size_t)HR_DIM * E_DIM * 2);
  u16* wqb   = (u16*)alloc((size_t)HR_DIM * HR_DIM * 2);
  u16* wkb   = (u16*)alloc((size_t)HR_DIM * HR_DIM * 2);
  u16* wvb   = (u16*)alloc((size_t)HR_DIM * HR_DIM * 2);
  u16* wf1b  = (u16*)alloc((size_t)2048 * HR_DIM * 2);
  u16* wf2b  = (u16*)alloc((size_t)E_DIM * 2048 * 2);
  u16* wqi1b = (u16*)alloc((size_t)HR_DIM * E_DIM * 2);
  u16* wki1b = (u16*)alloc((size_t)HR_DIM * E_DIM * 2);
  u16* wqi2b = (u16*)alloc((size_t)16384 * HR_DIM * 2);
  u16* wki2b = (u16*)alloc((size_t)16384 * HR_DIM * 2);
  u16* hr    = (u16*)alloc((size_t)L_SEQ * HR_DIM * 2);
  u16* qbuf  = (u16*)alloc((size_t)L_SEQ * HR_DIM * 2);
  u16* kbuf  = (u16*)alloc((size_t)L_SEQ * HR_DIM * 2);
  u16* vtbuf = (u16*)alloc((size_t)HR_DIM * L_SEQ * 2);
  u16* atb   = (u16*)alloc((size_t)L_SEQ * HR_DIM * 2);
  u16* lnb   = (u16*)alloc((size_t)L_SEQ * HR_DIM * 2);
  u16* f1b   = (u16*)alloc((size_t)L_SEQ * 2048 * 2);
  u16* hid   = (u16*)alloc((size_t)L_SEQ * E_DIM * 2);
  u16* siq   = (u16*)alloc((size_t)L_SEQ * HR_DIM * 2);
  u16* sik   = (u16*)alloc((size_t)L_SEQ * HR_DIM * 2);

  auto cvt = [&](const float* s, u16* d, size_t n) {
    int n4 = (int)(n / 4);
    cvt_f32_bf16<<<(n4 + 255) / 256, 256, 0, stream>>>(s, d, n4);
  };
  cvt(x,    xb,    (size_t)L_SEQ * E_DIM);
  cvt(Wi,   wib,   (size_t)HR_DIM * E_DIM);
  cvt(Wq,   wqb,   (size_t)HR_DIM * HR_DIM);
  cvt(Wk,   wkb,   (size_t)HR_DIM * HR_DIM);
  cvt(Wv,   wvb,   (size_t)HR_DIM * HR_DIM);
  cvt(Wf1,  wf1b,  (size_t)2048 * HR_DIM);
  cvt(Wf2,  wf2b,  (size_t)E_DIM * 2048);
  cvt(Wqi1, wqi1b, (size_t)HR_DIM * E_DIM);
  cvt(Wki1, wki1b, (size_t)HR_DIM * E_DIM);
  cvt(Wqi2, wqi2b, (size_t)16384 * HR_DIM);
  cvt(Wki2, wki2b, (size_t)16384 * HR_DIM);

  auto g = [](int N, int M) { return dim3((unsigned)(N / 128), (unsigned)(M / 128)); };

  // hr = x @ Wi^T
  gemm_bf16k<EPI_BF16><<<g(HR_DIM, L_SEQ), 256, 0, stream>>>(
      xb, wib, nullptr, nullptr, hr, L_SEQ, HR_DIM, E_DIM);
  // q, k, v
  gemm_bf16k<EPI_BF16><<<g(HR_DIM, L_SEQ), 256, 0, stream>>>(
      hr, wqb, nullptr, nullptr, qbuf, L_SEQ, HR_DIM, HR_DIM);
  gemm_bf16k<EPI_BF16><<<g(HR_DIM, L_SEQ), 256, 0, stream>>>(
      hr, wkb, nullptr, nullptr, kbuf, L_SEQ, HR_DIM, HR_DIM);
  gemm_bf16k<EPI_T><<<g(HR_DIM, L_SEQ), 256, 0, stream>>>(
      hr, wvb, nullptr, nullptr, vtbuf, L_SEQ, HR_DIM, HR_DIM);
  // rope on q, k (d=32)
  rope32_kernel<<<(L_SEQ * 512) / 256, 256, 0, stream>>>(qbuf);
  rope32_kernel<<<(L_SEQ * 512) / 256, 256, 0, stream>>>(kbuf);
  // causal flash attention
  attn_kernel<<<dim3(16, 32), 256, 0, stream>>>(qbuf, kbuf, vtbuf, atb);
  // layernorm
  ln_kernel<<<L_SEQ, 256, 0, stream>>>(atb, ln1w, ln1b, lnb);
  // ffn
  gemm_bf16k<EPI_BIAS_GELU><<<g(2048, L_SEQ), 256, 0, stream>>>(
      lnb, wf1b, bf1, nullptr, f1b, L_SEQ, 2048, HR_DIM);
  gemm_bf16k<EPI_BIAS_RESID><<<g(E_DIM, L_SEQ), 256, 0, stream>>>(
      f1b, wf2b, bf2v, x, hid, L_SEQ, E_DIM, 2048);
  // importance projections
  gemm_bf16k<EPI_SILU><<<g(HR_DIM, L_SEQ), 256, 0, stream>>>(
      hid, wqi1b, nullptr, nullptr, siq, L_SEQ, HR_DIM, E_DIM);
  gemm_bf16k<EPI_SILU><<<g(HR_DIM, L_SEQ), 256, 0, stream>>>(
      hid, wki1b, nullptr, nullptr, sik, L_SEQ, HR_DIM, E_DIM);
  // final GEMMs with fused rope(d=16) + permuted fp32 store
  gemm_bf16k<EPI_ROPE><<<g(16384, L_SEQ), 256, 0, stream>>>(
      siq, wqi2b, nullptr, nullptr, out, L_SEQ, 16384, HR_DIM);
  gemm_bf16k<EPI_ROPE><<<g(16384, L_SEQ), 256, 0, stream>>>(
      sik, wki2b, nullptr, nullptr, out + (size_t)33554432, L_SEQ, 16384, HR_DIM);
}
